// BertSelfAttention_16106127360509
// MI455X (gfx1250) — matrix operand order repaired and checked by reading, not compiled, not run on hardware
//
#include <hip/hip_runtime.h>

typedef __attribute__((ext_vector_type(16))) __bf16 v16bf;
typedef __attribute__((ext_vector_type(8)))  __bf16 v8bf;
typedef __attribute__((ext_vector_type(8)))  float  v8f;

union FragBF {
    v16bf v;
    v8bf  h[2];
};

// Problem constants (match reference)
static constexpr int CB = 4;      // batch
static constexpr int CS = 2048;   // seq
static constexpr int CD = 1024;   // model dim
static constexpr int CH = 16;     // heads
static constexpr int CHD = 64;    // head dim

// CDNA5 async global->LDS copy of 16 bytes (GV mode), tracked by ASYNCcnt.
// LDS address = low 32 bits of the generic shared pointer (flat aperture rule).
#define ASYNC_COPY_B128(ldsptr, gptr)                                          \
    do {                                                                       \
        unsigned _l = (unsigned)(uintptr_t)(ldsptr);                           \
        unsigned long long _g = (unsigned long long)(uintptr_t)(gptr);         \
        asm volatile("global_load_async_to_lds_b128 %0, %1, off"               \
                     :: "v"(_l), "v"(_g) : "memory");                          \
    } while (0)

// Wait until at most N of this wave's async copies remain outstanding.
#define WAIT_ASYNC(n) asm volatile("s_wait_asynccnt %0" :: "i"(n) : "memory")

// ---------------------------------------------------------------------------
// Kernel 1: fused QKV projection. Y = X @ W + b, stored bf16 as [B,H,S,HD].
// Block: 256 threads (8 waves). Block tile: 128(M) x 64(N), K-step 32.
// W tile staged COLUMN-major in LDS so B-fragments are contiguous b128 loads.
// All 4 B-fragments preloaded, then 4 back-to-back WMMAs.
// grid = (M/128, N/64, 3) with z selecting {Q,K,V}.
// ---------------------------------------------------------------------------
__global__ __launch_bounds__(256)
void qkv_gemm_kernel(const float* __restrict__ X,
                     const float* __restrict__ Wq, const float* __restrict__ bq,
                     const float* __restrict__ Wk, const float* __restrict__ bk,
                     const float* __restrict__ Wv, const float* __restrict__ bv,
                     __bf16* __restrict__ q_ws, __bf16* __restrict__ k_ws,
                     __bf16* __restrict__ v_ws)
{
    __shared__ __align__(16) __bf16 lX [128 * 32];  // [m][k]  row-major
    __shared__ __align__(16) __bf16 lWt[64 * 32];   // [n][k]  column-major(!)

    const int tid  = threadIdx.x;
    const int wv   = tid >> 5;
    const int lane = tid & 31;
    const int l15  = lane & 15;
    const int half = lane >> 4;

    const float* Wm;
    const float* bias;
    __bf16* outw;
    if (blockIdx.z == 0)      { Wm = Wq; bias = bq; outw = q_ws; }
    else if (blockIdx.z == 1) { Wm = Wk; bias = bk; outw = k_ws; }
    else                      { Wm = Wv; bias = bv; outw = v_ws; }

    const int m0 = blockIdx.x * 128;
    const int n0 = blockIdx.y * 64;

    v8f acc[4] = {v8f{}, v8f{}, v8f{}, v8f{}};

    for (int kc = 0; kc < CD / 32; ++kc) {
        const int k0 = kc * 32;
        __syncthreads();

        {   // Stage X tile 128x32 -> bf16, row-major (contiguous stores)
            const int r  = tid >> 1;
            const int ko = (tid & 1) * 16;
            const float* src = X + (size_t)(m0 + r) * CD + k0 + ko;
#pragma unroll
            for (int i = 0; i < 16; ++i) lX[r * 32 + ko + i] = (__bf16)src[i];
        }
        {   // Stage W tile 32x64 -> bf16, TRANSPOSED to [n][k]
            const int r  = tid >> 3;          // k row 0..31
            const int no = (tid & 7) * 8;     // n col group
            const float* src = Wm + (size_t)(k0 + r) * CD + n0 + no;
#pragma unroll
            for (int i = 0; i < 8; ++i) lWt[(no + i) * 32 + r] = (__bf16)src[i];
        }
        __syncthreads();

        // A fragment 16x32: lane = row m; two contiguous 8-elem runs
        FragBF a;
        {
            const int m = wv * 16 + l15;
            a.h[0] = *(const v8bf*)&lX[m * 32 + half * 8];
            a.h[1] = *(const v8bf*)&lX[m * 32 + 16 + half * 8];
        }
        // Preload all 4 B fragments (two 8-elem runs, same k order as A)
        FragBF bfr[4];
#pragma unroll
        for (int g = 0; g < 4; ++g) {
            const int n = g * 16 + l15;
            bfr[g].h[0] = *(const v8bf*)&lWt[n * 32 + half * 8];
            bfr[g].h[1] = *(const v8bf*)&lWt[n * 32 + 16 + half * 8];
        }
        // Back-to-back WMMAs (independent accumulators)
#pragma unroll
        for (int g = 0; g < 4; ++g)
            acc[g] = __builtin_amdgcn_wmma_f32_16x16x32_bf16(
                false, a.v, false, bfr[g].v, (short)0, acc[g], false, false);
    }

    // Epilogue: bias add, bf16 convert, scatter to [B,H,S,HD]
#pragma unroll
    for (int g = 0; g < 4; ++g) {
        const int n     = n0 + g * 16 + l15;
        const float bia = bias[n];
        const int hh = n >> 6;
        const int hd = n & 63;
#pragma unroll
        for (int r = 0; r < 8; ++r) {
            const int m = m0 + wv * 16 + r + half * 8;
            const int b = m >> 11;
            const int s = m & 2047;
            outw[((size_t)((b * CH + hh) * CS + s)) * CHD + hd] =
                (__bf16)(acc[g][r] + bia);
        }
    }
}

// ---------------------------------------------------------------------------
// Kernel 2: flash attention over bf16 Q/K/V in [B,H,S,HD].
// Block: 256 threads (8 waves) handles 128 query rows of one (b,h).
// K/V staged with GLOBAL_LOAD_ASYNC_TO_LDS_B128 into DOUBLE-BUFFERED tiles:
// next tile's DMA overlaps current tile's WMMA/softmax (s_wait_asynccnt 2).
// V transposed in-LDS so its B-fragments are contiguous b128 loads.
// grid = (S/128, B*H).
// ---------------------------------------------------------------------------
__global__ __launch_bounds__(256)
void attn_kernel(const __bf16* __restrict__ q_ws,
                 const __bf16* __restrict__ k_ws,
                 const __bf16* __restrict__ v_ws,
                 const float* __restrict__ mask,
                 float* __restrict__ out)
{
    __shared__ __align__(16) __bf16 lQ   [128 * 64];     // 16 KB  [q][d]
    __shared__ __align__(16) __bf16 lK   [2][32 * 64];   //  8 KB  [key][d] x2
    __shared__ __align__(16) __bf16 lVrow[2][32 * 64];   //  8 KB  [key][d] x2
    __shared__ __align__(16) __bf16 lVt  [64 * 32];      //  4 KB  [d][key]
    __shared__ __align__(16) __bf16 lP[8][16 * 32];      //  8 KB  per-wave P

    const int tid  = threadIdx.x;
    const int wv   = tid >> 5;
    const int lane = tid & 31;
    const int l15  = lane & 15;
    const int half = lane >> 4;

    const int q0 = blockIdx.x * 128;
    const int bh = blockIdx.y;
    const int b  = bh >> 4;
    const int h  = bh & 15;
    const size_t headbase = (size_t)bh * CS * CHD;

    // Per-thread K/V staging coordinates: 16B per thread per tile
    const int sr    = tid >> 3;          // key row 0..31
    const int sdoff = (tid & 7) * 8;     // dim offset

    {   // Async-stage Q tile (128x64 bf16 = 16 KB): 4 x 16B per thread
        const int r    = tid >> 1;
        const int doff = (tid & 1) * 32;
        const __bf16* src = q_ws + headbase + (size_t)(q0 + r) * CHD + doff;
#pragma unroll
        for (int i = 0; i < 4; ++i)
            ASYNC_COPY_B128(&lQ[r * 64 + doff + i * 8], src + i * 8);
    }
    // Prologue: issue iteration-0 K/V tiles (2 copies/thread), buffer 0
    {
        const __bf16* ks = k_ws + headbase + (size_t)sr * CHD + sdoff;
        const __bf16* vs = v_ws + headbase + (size_t)sr * CHD + sdoff;
        ASYNC_COPY_B128(&lK[0][sr * 64 + sdoff], ks);
        ASYNC_COPY_B128(&lVrow[0][sr * 64 + sdoff], vs);
    }
    WAIT_ASYNC(2);        // Q resident (iter-0 K/V may still be in flight)
    __syncthreads();

    // Q A-fragments: two 16x32 chunks over HD=64, contiguous 8-elem runs
    FragBF qf[2];
#pragma unroll
    for (int c = 0; c < 2; ++c) {
        const int m = wv * 16 + l15;
        qf[c].h[0] = *(const v8bf*)&lQ[m * 64 + c * 32 + half * 8];
        qf[c].h[1] = *(const v8bf*)&lQ[m * 64 + c * 32 + 16 + half * 8];
    }

    v8f O[4] = {v8f{}, v8f{}, v8f{}, v8f{}};
    float rm[8], rl[8];
#pragma unroll
    for (int r = 0; r < 8; ++r) { rm[r] = -1e30f; rl[r] = 0.0f; }

    const float sc  = 0.125f;            // 1/sqrt(HD)
    const float l2e = 1.44269504088896f; // log2(e)

    constexpr int NIT = CS / 32;         // 64 key-block iterations
    for (int it = 0; it < NIT; ++it) {
        const int cur = it & 1;
        const int kb0 = it * 32;
        __syncthreads();   // everyone done with buffers from iteration it-1

        if (it + 1 < NIT) {
            // Issue next tile's async copies into the other buffer: DMA
            // overlaps this iteration's compute.
            const int nb0 = (it + 1) * 32;
            const __bf16* ks = k_ws + headbase + (size_t)(nb0 + sr) * CHD + sdoff;
            const __bf16* vs = v_ws + headbase + (size_t)(nb0 + sr) * CHD + sdoff;
            ASYNC_COPY_B128(&lK[1 - cur][sr * 64 + sdoff], ks);
            ASYNC_COPY_B128(&lVrow[1 - cur][sr * 64 + sdoff], vs);
            WAIT_ASYNC(2); // in-order: current buffer's 2 copies have landed
        } else {
            WAIT_ASYNC(0);
        }
        __syncthreads();   // all waves' current-tile copies visible

        {   // In-LDS transpose of V: [key][d] -> [d][key]
            const v8bf row = *(const v8bf*)&lVrow[cur][sr * 64 + sdoff];
#pragma unroll
            for (int i = 0; i < 8; ++i) lVt[(sdoff + i) * 32 + sr] = row[i];
        }

        // Scores: preload 4 K^T B-fragments, then 4 back-to-back WMMAs
        FragBF kf[2][2];
#pragma unroll
        for (int t = 0; t < 2; ++t)
#pragma unroll
            for (int c = 0; c < 2; ++c) {
                const int nk = t * 16 + l15;
                kf[t][c].h[0] =
                    *(const v8bf*)&lK[cur][nk * 64 + c * 32 + half * 8];
                kf[t][c].h[1] =
                    *(const v8bf*)&lK[cur][nk * 64 + c * 32 + 16 + half * 8];
            }
        v8f sf[2] = {v8f{}, v8f{}};
#pragma unroll
        for (int t = 0; t < 2; ++t)
#pragma unroll
            for (int c = 0; c < 2; ++c)
                sf[t] = __builtin_amdgcn_wmma_f32_16x16x32_bf16(
                    false, qf[c].v, false, kf[t][c].v, (short)0, sf[t],
                    false, false);

        const float mv0 = mask[b * CS + kb0 + l15];
        const float mv1 = mask[b * CS + kb0 + 16 + l15];
#pragma unroll
        for (int r = 0; r < 8; ++r) {
            sf[0][r] = sf[0][r] * sc + mv0;
            sf[1][r] = sf[1][r] * sc + mv1;
        }

        // Online softmax: row stats via 16-lane xor-shuffle reductions
        float nm[8], al[8], p0[8], p1[8];
#pragma unroll
        for (int r = 0; r < 8; ++r) {
            float v = fmaxf(sf[0][r], sf[1][r]);
#pragma unroll
            for (int ms = 8; ms >= 1; ms >>= 1)
                v = fmaxf(v, __shfl_xor(v, ms, 16));
            nm[r] = fmaxf(rm[r], v);
            al[r] = __builtin_amdgcn_exp2f((rm[r] - nm[r]) * l2e);
        }
#pragma unroll
        for (int r = 0; r < 8; ++r) {
            p0[r] = __builtin_amdgcn_exp2f((sf[0][r] - nm[r]) * l2e);
            p1[r] = __builtin_amdgcn_exp2f((sf[1][r] - nm[r]) * l2e);
            float s = p0[r] + p1[r];
#pragma unroll
            for (int ms = 8; ms >= 1; ms >>= 1)
                s += __shfl_xor(s, ms, 16);
            rl[r] = rl[r] * al[r] + s;
            rm[r] = nm[r];
        }

        // Stage P (C layout) to LDS for A-fragment reload
#pragma unroll
        for (int r = 0; r < 8; ++r) {
            const int m = r + half * 8;
            lP[wv][m * 32 + l15]      = (__bf16)p0[r];
            lP[wv][m * 32 + 16 + l15] = (__bf16)p1[r];
        }
        __syncthreads();   // orders lP + lVt writes before fragment reads

        // Rescale running context by alpha
#pragma unroll
        for (int g = 0; g < 4; ++g)
#pragma unroll
            for (int r = 0; r < 8; ++r) O[g][r] *= al[r];

        // P A-fragment (16 queries x 32 keys): two contiguous 8-elem runs
        FragBF pf;
        pf.h[0] = *(const v8bf*)&lP[wv][l15 * 32 + half * 8];
        pf.h[1] = *(const v8bf*)&lP[wv][l15 * 32 + 16 + half * 8];

        // Preload 4 V B-fragments, then 4 back-to-back WMMAs
        FragBF vf[4];
#pragma unroll
        for (int g = 0; g < 4; ++g) {
            const int nd = g * 16 + l15;
            vf[g].h[0] = *(const v8bf*)&lVt[nd * 32 + half * 8];
            vf[g].h[1] = *(const v8bf*)&lVt[nd * 32 + 16 + half * 8];
        }
#pragma unroll
        for (int g = 0; g < 4; ++g)
            O[g] = __builtin_amdgcn_wmma_f32_16x16x32_bf16(
                false, pf.v, false, vf[g].v, (short)0, O[g], false, false);
    }

    // Normalize and store fp32 output [B,S,D]
#pragma unroll
    for (int r = 0; r < 8; ++r) {
        const float inv = 1.0f / rl[r];
        const int q = q0 + wv * 16 + r + half * 8;
#pragma unroll
        for (int g = 0; g < 4; ++g) {
            const int dcol = g * 16 + l15;
            out[(size_t)(b * CS + q) * CD + h * CHD + dcol] = O[g][r] * inv;
        }
    }
}

// ---------------------------------------------------------------------------
extern "C" void kernel_launch(void* const* d_in, const int* in_sizes, int n_in,
                              void* d_out, int out_size, void* d_ws, size_t ws_size,
                              hipStream_t stream)
{
    const float* X    = (const float*)d_in[0];
    const float* mask = (const float*)d_in[1];
    const float* Wq   = (const float*)d_in[2];
    const float* bq   = (const float*)d_in[3];
    const float* Wk   = (const float*)d_in[4];
    const float* bk   = (const float*)d_in[5];
    const float* Wv   = (const float*)d_in[6];
    const float* bv   = (const float*)d_in[7];
    float* out = (float*)d_out;

    // Workspace: Q,K,V bf16 [B,H,S,HD], 16 MB each (48 MB total)
    const size_t per = (size_t)CB * CS * CD;
    __bf16* q_ws = (__bf16*)d_ws;
    __bf16* k_ws = q_ws + per;
    __bf16* v_ws = k_ws + per;

    dim3 g1(CB * CS / 128, CD / 64, 3);   // (64, 16, 3)
    qkv_gemm_kernel<<<g1, 256, 0, stream>>>(X, Wq, bq, Wk, bk, Wv, bv,
                                            q_ws, k_ws, v_ws);

    dim3 g2(CS / 128, CB * CH);           // (16, 64)
    attn_kernel<<<g2, 256, 0, stream>>>(q_ws, k_ws, v_ws, mask, out);
}
